// EfficientMoELayer_30459908063734
// MI455X (gfx1250) — matrix operand order repaired and checked by reading, not compiled, not run on hardware
//
#include <hip/hip_runtime.h>
#include <hip/hip_bf16.h>

#define TOK_N 16384
#define DIM_H 1024
#define DIM_F 4096
#define NEXP  8
#define MT    32

#define XT_STR  1032
#define WB_STR  34
#define WC_STR  130
#define HC_STR  136

typedef __attribute__((ext_vector_type(16))) __bf16 v16bf;
typedef __attribute__((ext_vector_type(8)))  float  v8f;

union FragBF { v16bf v; unsigned int u[8]; };

// pack two f32 -> two bf16 in one dword (v_cvt_pk_bf16_f32)
__device__ __forceinline__ unsigned int pack_bf2(float a, float b) {
  union { __bf16 h[2]; unsigned int u; } p;
  p.h[0] = (__bf16)a;
  p.h[1] = (__bf16)b;
  return p.u;
}

__device__ __forceinline__ unsigned short f2bf(float f) {
  union { __bf16 h[2]; unsigned short s[2]; } p;
  p.h[0] = (__bf16)f;
  return p.s[0];
}

// gelu(tanh approx) == x * sigmoid(2*z), z = sqrt(2/pi)*(x + 0.044715 x^3)
// branch-free: v_exp_f32 + v_rcp_f32 (avoid the IEEE div_scale/div_fmas sequence)
__device__ __forceinline__ float gelu_tanh(float x) {
  float z = 0.7978845608028654f * (x + 0.044715f * x * x * x);
  return x * __builtin_amdgcn_rcpf(1.0f + __expf(-2.0f * z));
}

// A fragment (16x32 bf16), ISA 7.12.2: lane L row = L%16; VGPR v holds K pair
__device__ __forceinline__ v16bf load_a_frag(const unsigned short* base, int stride,
                                             int row, int kbase, int lane) {
  FragBF f;
  const unsigned short* p = base + (row + (lane & 15)) * stride + kbase + ((lane >> 4) << 3);
#pragma unroll
  for (int v = 0; v < 8; ++v) {
    int koff = ((v & 3) << 1) + ((v >> 2) << 4);
    f.u[v] = *(const unsigned int*)(p + koff);
  }
  return f.v;
}

// B fragment (32x16 bf16) from LDS tile stored transposed [n][k]
__device__ __forceinline__ v16bf load_b_frag(const unsigned short* base, int stride,
                                             int col, int kbase, int lane) {
  FragBF f;
  const unsigned short* p = base + (col + (lane & 15)) * stride + kbase + ((lane >> 4) << 3);
#pragma unroll
  for (int v = 0; v < 8; ++v)
    f.u[v] = *(const unsigned int*)(p + ((v & 3) << 1) + ((v >> 2) << 4));
  return f.v;
}

// ---------------- Router: one wave (32 lanes) per token ----------------
__global__ void __launch_bounds__(256, 4)
moe_router(const float* __restrict__ x, const float* __restrict__ Wg,
           int* __restrict__ counts, int* __restrict__ tok_ids, float* __restrict__ tok_w) {
  const int lane = threadIdx.x & 31;
  const int n = blockIdx.x * 8 + (threadIdx.x >> 5);
  const float* xr = x + (size_t)n * DIM_H;
  float acc[NEXP];
#pragma unroll
  for (int e = 0; e < NEXP; ++e) acc[e] = 0.0f;
  for (int h = lane; h < DIM_H; h += 32) {
    float xv = xr[h];
    const float4* g = (const float4*)(Wg + h * NEXP);
    float4 g0 = g[0], g1 = g[1];
    acc[0] += xv * g0.x; acc[1] += xv * g0.y; acc[2] += xv * g0.z; acc[3] += xv * g0.w;
    acc[4] += xv * g1.x; acc[5] += xv * g1.y; acc[6] += xv * g1.z; acc[7] += xv * g1.w;
  }
#pragma unroll
  for (int e = 0; e < NEXP; ++e) {
#pragma unroll
    for (int off = 16; off > 0; off >>= 1)
      acc[e] += __shfl_xor(acc[e], off, 32);
  }
  if (lane == 0) {
    int i0 = 0;
#pragma unroll
    for (int e = 1; e < NEXP; ++e) if (acc[e] > acc[i0]) i0 = e;
    int i1 = (i0 == 0) ? 1 : 0;
#pragma unroll
    for (int e = 0; e < NEXP; ++e) if (e != i0 && acc[e] > acc[i1]) i1 = e;
    float w1v = __expf(acc[i1] - acc[i0]);   // softmax over the two selected logits
    float s = 1.0f + w1v;
    int p0 = atomicAdd(&counts[i0], 1);
    tok_ids[i0 * TOK_N + p0] = n;
    tok_w  [i0 * TOK_N + p0] = 1.0f / s;
    int p1 = atomicAdd(&counts[i1], 1);
    tok_ids[i1 * TOK_N + p1] = n;
    tok_w  [i1 * TOK_N + p1] = w1v / s;
  }
}

// ------------- Fused expert FFN: 32 gathered tokens x 1 expert per block -------------
// LDS ~155 KB -> two workgroups per 320 KB WGP. Output tile lives in registers
// (128 VGPRs/lane; gfx1250 extended VGPR file via s_set_vgpr_msb).
__global__ void __launch_bounds__(256)
moe_expert_ffn(const float* __restrict__ x,
               const float* __restrict__ W1, const float* __restrict__ b1,
               const float* __restrict__ W2, const float* __restrict__ b2,
               const int* __restrict__ counts, const int* __restrict__ tok_ids,
               const float* __restrict__ tok_w, float* __restrict__ out) {
  __shared__ unsigned short Xt[MT * XT_STR];        // 64.5 KB bf16 X tile (gathered)
  __shared__ unsigned short Wb[2][128 * WB_STR];    // 17 KB   W1 k-slab x2, [f][k]
  __shared__ unsigned short Wc[2][128 * WC_STR];    // 65 KB   W2 slab x2,   [n][k]
  __shared__ unsigned short Hc[MT * HC_STR];        // 8.5 KB  gelu(h) chunk, bf16
  __shared__ int   s_tok[MT];
  __shared__ float s_w[MT];

  const int e   = blockIdx.y;
  const int cnt = counts[e];
  const int m0  = blockIdx.x * MT;
  if (m0 >= cnt) return;                            // block-uniform
  const int mrows = min(MT, cnt - m0);

  const int tid  = threadIdx.x;
  const int lane = tid & 31;
  const int wv   = tid >> 5;

  if (tid < MT) {
    s_tok[tid] = (tid < mrows) ? tok_ids[e * TOK_N + m0 + tid] : 0;
    s_w[tid]   = (tid < mrows) ? tok_w[e * TOK_N + m0 + tid]   : 0.0f;
  }
  __syncthreads();

  for (int i = tid; i < MT * (DIM_H / 2); i += 256) {  // pack 2 cols per store
    int r = i >> 9, h = (i & 511) << 1;
    float xa = 0.0f, xb = 0.0f;
    if (r < mrows) {
      const float2 xv = *(const float2*)(x + (size_t)s_tok[r] * DIM_H + h);
      xa = xv.x; xb = xv.y;
    }
    *(unsigned int*)&Xt[r * XT_STR + h] = pack_bf2(xa, xb);
  }

  const float* W1e = W1 + (size_t)e * DIM_H * DIM_F;
  const float* W2e = W2 + (size_t)e * DIM_F * DIM_H;

  const int m_sub = wv & 1;            // 2 row subtiles of 16
  const int n0    = (wv >> 1) << 1;    // 2 of 8 col subtiles per 128-wide chunk
  const int laneHalf = lane >> 4, lcol = lane & 15;

  v8f cacc[16];                        // output 32x1024: 16 C-tiles per wave, in VGPRs
#pragma unroll
  for (int t = 0; t < 16; ++t) cacc[t] = (v8f){};

  for (int fc = 0; fc < DIM_F; fc += 128) {
    // ---- GEMM1: Hc[32x128] = gelu(Xt[32x1024] @ W1[:, fc:fc+128] + b1) ----
    v8f acc0 = {}; v8f acc1 = {};
    for (int kk = 0; kk < DIM_H; kk += 32) {
      const int buf = (kk >> 5) & 1;
      { // stage W1[kk:kk+32, fc:fc+128] -> Wb[buf][f][k] (bf16, transposed, paired k)
        int k0 = (tid >> 4) << 1;              // 0..30 even
        int f0 = (tid & 15) << 3;              // 0..120 step 8
        const float4* r0 = (const float4*)(W1e + (size_t)(kk + k0) * DIM_F + fc + f0);
        const float4* r1 = (const float4*)(W1e + (size_t)(kk + k0 + 1) * DIM_F + fc + f0);
        float4 a0 = r0[0], a1 = r0[1], b0v = r1[0], b1v = r1[1];
        const float av[8] = {a0.x, a0.y, a0.z, a0.w, a1.x, a1.y, a1.z, a1.w};
        const float bv[8] = {b0v.x, b0v.y, b0v.z, b0v.w, b1v.x, b1v.y, b1v.z, b1v.w};
#pragma unroll
        for (int i = 0; i < 8; ++i)
          *(unsigned int*)&Wb[buf][(f0 + i) * WB_STR + k0] = pack_bf2(av[i], bv[i]);
      }
      __syncthreads();   // waits own dscnt first; other buffer still readable by laggards
      v16bf a    = load_a_frag(Xt, XT_STR, m_sub * 16, kk, lane);
      v16bf bfr0 = load_b_frag(Wb[buf], WB_STR, n0 * 16, 0, lane);
      v16bf bfr1 = load_b_frag(Wb[buf], WB_STR, (n0 + 1) * 16, 0, lane);
      acc0 = __builtin_amdgcn_wmma_f32_16x16x32_bf16(false, a, false, bfr0, (short)0, acc0, false, false);
      acc1 = __builtin_amdgcn_wmma_f32_16x16x32_bf16(false, a, false, bfr1, (short)0, acc1, false, false);
    }
    { // bias + gelu, write bf16 Hc tile (C/D layout: m = r + 8*(lane/16), n = lane%16)
#pragma unroll
      for (int j = 0; j < 2; ++j) {
        int nn = (n0 + j) * 16 + lcol;
        float bias = b1[e * DIM_F + fc + nn];
        v8f accv = j ? acc1 : acc0;
#pragma unroll
        for (int r = 0; r < 8; ++r) {
          int m = m_sub * 16 + r + (laneHalf << 3);
          Hc[m * HC_STR + nn] = f2bf(gelu_tanh(accv[r] + bias));
        }
      }
    }
    __syncthreads();     // Hc writes (all waves) before hoisted A-fragment reads

    // A-fragments of Hc are ncb-invariant: load the 4 k-step fragments ONCE per fc
    v16bf ha[4];
#pragma unroll
    for (int ks = 0; ks < 4; ++ks)
      ha[ks] = load_a_frag(Hc, HC_STR, m_sub * 16, ks * 32, lane);

    // ---- GEMM2: cacc[32x1024] += Hc[32x128] @ W2[fc:fc+128, :] ----
#pragma unroll
    for (int ncb = 0; ncb < 8; ++ncb) {
      const int nc = ncb << 7;
      const int buf = ncb & 1;
      { // stage W2[fc:fc+128, nc:nc+128] -> Wc[buf][n][k] (bf16, transposed, paired k)
        int k0  = (tid >> 2) << 1;             // 0..126 even
        int nb  = (tid & 3) << 5;              // 0,32,64,96
        const float4* r0 = (const float4*)(W2e + (size_t)(fc + k0) * DIM_H + nc + nb);
        const float4* r1 = (const float4*)(W2e + (size_t)(fc + k0 + 1) * DIM_H + nc + nb);
#pragma unroll
        for (int q = 0; q < 8; ++q) {
          float4 va = r0[q], vb = r1[q];
          int nn = nb + (q << 2);
          *(unsigned int*)&Wc[buf][(nn + 0) * WC_STR + k0] = pack_bf2(va.x, vb.x);
          *(unsigned int*)&Wc[buf][(nn + 1) * WC_STR + k0] = pack_bf2(va.y, vb.y);
          *(unsigned int*)&Wc[buf][(nn + 2) * WC_STR + k0] = pack_bf2(va.z, vb.z);
          *(unsigned int*)&Wc[buf][(nn + 3) * WC_STR + k0] = pack_bf2(va.w, vb.w);
        }
      }
      __syncthreads();
#pragma unroll
      for (int ks = 0; ks < 4; ++ks) {
        v16bf bfr0 = load_b_frag(Wc[buf], WC_STR, n0 * 16, ks * 32, lane);
        v16bf bfr1 = load_b_frag(Wc[buf], WC_STR, (n0 + 1) * 16, ks * 32, lane);
        cacc[ncb * 2 + 0] = __builtin_amdgcn_wmma_f32_16x16x32_bf16(
            false, ha[ks], false, bfr0, (short)0, cacc[ncb * 2 + 0], false, false);
        cacc[ncb * 2 + 1] = __builtin_amdgcn_wmma_f32_16x16x32_bf16(
            false, ha[ks], false, bfr1, (short)0, cacc[ncb * 2 + 1], false, false);
      }
    }
  }

  // scatter from C fragments: out[token] += gate_w * (acc + b2)
  // exactly 2 commutative f32 adds per element -> deterministic
#pragma unroll
  for (int ncb = 0; ncb < 8; ++ncb) {
#pragma unroll
    for (int j = 0; j < 2; ++j) {
      int col = (ncb << 7) + (n0 + j) * 16 + lcol;
      float bias = b2[e * DIM_H + col];
      v8f cv = cacc[ncb * 2 + j];
#pragma unroll
      for (int r = 0; r < 8; ++r) {
        int m = m_sub * 16 + r + (laneHalf << 3);
        if (m < mrows) {
          float val = (cv[r] + bias) * s_w[m];
          unsafeAtomicAdd(&out[(size_t)s_tok[m] * DIM_H + col], val);
        }
      }
    }
  }
}

extern "C" void kernel_launch(void* const* d_in, const int* in_sizes, int n_in,
                              void* d_out, int out_size, void* d_ws, size_t ws_size,
                              hipStream_t stream) {
  const float* x  = (const float*)d_in[0];
  const float* Wg = (const float*)d_in[1];
  const float* W1 = (const float*)d_in[2];
  const float* b1 = (const float*)d_in[3];
  const float* W2 = (const float*)d_in[4];
  const float* b2 = (const float*)d_in[5];
  float* out = (float*)d_out;

  int*   counts  = (int*)d_ws;
  int*   tok_ids = (int*)((char*)d_ws + 256);
  float* tok_w   = (float*)((char*)d_ws + 256 + (size_t)NEXP * TOK_N * sizeof(int));

  hipMemsetAsync(counts, 0, 256, stream);
  hipMemsetAsync(out, 0, (size_t)out_size * sizeof(float), stream);

  moe_router<<<dim3(TOK_N / 8), dim3(256), 0, stream>>>(x, Wg, counts, tok_ids, tok_w);
  moe_expert_ffn<<<dim3(TOK_N / MT, NEXP), dim3(256), 0, stream>>>(
      x, W1, b1, W2, b2, counts, tok_ids, tok_w, out);
}
